// GNNClassifier_81776177316260
// MI455X (gfx1250) — matrix operand order repaired and checked by reading, not compiled, not run on hardware
//
#include <hip/hip_runtime.h>
#include <hip/hip_bf16.h>
#include <math.h>

// Problem constants (match reference)
#define NN   50000
#define EEDG 800000
#define DIN  128
#define EDIM 101
#define NHEAD 4
#define NCH  32
#define HCF  128          // NHEAD * NCH
#define ETOT (EEDG + NN)  // edges + self loops
#define NEG_SLOPE 0.2f

// ordinal encoding of -inf (0xFF800000 has sign bit -> ~u)
static constexpr unsigned ORD_NEG_INF = 0x007FFFFFu;

typedef __attribute__((ext_vector_type(16))) __bf16 v16bf;
typedef __attribute__((ext_vector_type(8)))  float  v8f;

// ---------- helpers ----------
__device__ __forceinline__ __bf16 f2bf(float f) {
  unsigned u = __float_as_uint(f);
  unsigned r = (u + 0x7FFFu + ((u >> 16) & 1u)) >> 16;   // RNE
  unsigned short s = (unsigned short)r;
  __bf16 b;
  __builtin_memcpy(&b, &s, 2);
  return b;
}
// order-preserving float <-> uint encoding for atomicMax on floats
__device__ __forceinline__ unsigned f2ord(float f) {
  unsigned u = __float_as_uint(f);
  return (u & 0x80000000u) ? ~u : (u | 0x80000000u);
}
__device__ __forceinline__ float ord2f(unsigned u) {
  return (u & 0x80000000u) ? __uint_as_float(u & 0x7FFFFFFFu) : __uint_as_float(~u);
}

// ---------- fills ----------
__global__ void fill_f32(float* p, float v, long long n) {
  long long i = (long long)blockIdx.x * blockDim.x + threadIdx.x;
  long long st = (long long)gridDim.x * blockDim.x;
  for (; i < n; i += st) p[i] = v;
}
__global__ void fill_u32(unsigned* p, unsigned v, long long n) {
  long long i = (long long)blockIdx.x * blockDim.x + threadIdx.x;
  long long st = (long long)gridDim.x * blockDim.x;
  for (; i < n; i += st) p[i] = v;
}

// ---------- WMMA GEMM: C[M,Nc] = act(A[M,K] @ W[K,Nc] + bias) ----------
// block = 256 threads (8 waves); block tile = 64 rows x Nc cols (Nc<=128).
// Wave w owns N-tile w*16 and iterates 4 M-subtiles (4 accumulators), so each
// B stage feeds 4 WMMAs/wave. A (64x32) and B-transposed (Ncx32) are staged in
// LDS as bf16 (converted once, RNE), so frag builds are pure ds loads laid out
// per the CDNA5 16x16x32 bf16 VGPR mappings:
//   A: lane<16 -> M=lane; VGPR v: K = (v>>2)*16 + half*8 + (v&3)*2 + {0,1}
//   B: N = lane&15;        VGPR v: K = (v>>2)*16 + half*8 + (v&3)*2 + {0,1}
//   C: VGPR r -> M = r + 8*half, N = lane&15
__global__ void gemm_wmma_bf16(const float* __restrict__ A, const float* __restrict__ W,
                               const float* __restrict__ bias, float* __restrict__ Cmat,
                               int M, int K, int Nc, int relu) {
  __shared__ __bf16 As[64 * 34];    // [row][k], stride 34 (even -> b32-aligned pairs)
  __shared__ __bf16 Bst[128 * 34];  // [n][k] transposed, stride 34
  const int tid  = threadIdx.x;
  const int wave = tid >> 5;
  const int lane = tid & 31;
  const int half = lane >> 4;
  const int l16  = lane & 15;
  const int m0   = blockIdx.x * 64;
  const int nt0  = wave * 16;

  v8f acc[4] = {v8f{}, v8f{}, v8f{}, v8f{}};
  const int kTiles = (K + 31) / 32;
  for (int kt = 0; kt < kTiles; ++kt) {
    const int k0 = kt * 32;
    // prefetch next k-tile while we stage/compute this one (global_prefetch_b8)
    if (kt + 1 < kTiles) {
      int pm = m0 + (tid & 63);
      if (pm < M) __builtin_prefetch(A + (long long)pm * K + k0 + 32, 0, 3);
      int pk = k0 + 32 + (tid & 31);
      if (pk < K && tid < 32) __builtin_prefetch(W + (long long)pk * Nc, 0, 3);
    }
    // stage A 64x32 as bf16 (convert once; all 8 waves reuse)
    for (int i = tid; i < 64 * 32; i += 256) {
      int r = i >> 5, c = i & 31;
      int m = m0 + r, k = k0 + c;
      As[r * 34 + c] = f2bf((m < M && k < K) ? A[(long long)m * K + k] : 0.0f);
    }
    // stage B strip 32 x Nc as bf16, transposed -> Bst[n][k]
    for (int i = tid; i < 32 * Nc; i += 256) {
      int r = i / Nc, c = i - r * Nc;  // r = k within tile, c = n
      int k = k0 + r;
      Bst[c * 34 + r] = f2bf((k < K) ? W[(long long)k * Nc + c] : 0.0f);
    }
    __syncthreads();
    if (nt0 < Nc) {
      v16bf bf;
#pragma unroll
      for (int v = 0; v < 8; ++v) {
        int kb = ((v >> 2) * 16) + half * 8 + ((v & 3) * 2);
        bf[2 * v + 0] = Bst[(nt0 + l16) * 34 + kb + 0];
        bf[2 * v + 1] = Bst[(nt0 + l16) * 34 + kb + 1];
      }
#pragma unroll
      for (int ms = 0; ms < 4; ++ms) {
        v16bf af;
        const int mr = ms * 16 + l16;
#pragma unroll
        for (int v = 0; v < 8; ++v) {
          int ka = ((v >> 2) * 16) + half * 8 + ((v & 3) * 2);
          af[2 * v + 0] = As[mr * 34 + ka + 0];
          af[2 * v + 1] = As[mr * 34 + ka + 1];
        }
        acc[ms] = __builtin_amdgcn_wmma_f32_16x16x32_bf16(false, af, false, bf,
                                                          (short)0, acc[ms], false, false);
      }
    }
    __syncthreads();
  }
  if (nt0 < Nc) {
    const int nc = nt0 + l16;
#pragma unroll
    for (int ms = 0; ms < 4; ++ms) {
#pragma unroll
      for (int r = 0; r < 8; ++r) {
        int m = m0 + ms * 16 + r + 8 * half;
        if (m < M) {
          float v = acc[ms][r];
          if (bias) v += bias[nc];
          if (relu) v = fmaxf(v, 0.0f);
          Cmat[(long long)m * Nc + nc] = v;
        }
      }
    }
  }
}

// ---------- self-loop attr: per-dst degree + edge_attr sum ----------
__global__ void deg_loopsum(const long long* __restrict__ ei, const float* __restrict__ ea,
                            float* __restrict__ cnt, float* __restrict__ loopsum) {
  int e = blockIdx.x;
  if (e >= EEDG) return;
  int dst = (int)ei[EEDG + e];
  int t = threadIdx.x;
  if (t == 0) atomicAdd(&cnt[dst], 1.0f);
  if (t < EDIM)
    atomicAdd(&loopsum[(long long)dst * EDIM + t], ea[(long long)e * EDIM + t]);
}
__global__ void loopattr_fin(float* __restrict__ loopsum, const float* __restrict__ cnt) {
  int n = blockIdx.x, t = threadIdx.x;
  if (t < EDIM) loopsum[(long long)n * EDIM + t] /= fmaxf(cnt[n], 1.0f);
}

// ---------- GATv2 edge passes (wave per edge, 4 feats/lane, head = lane>>3) ----------
__global__ void edge_logit_k(const long long* __restrict__ ei, const float* __restrict__ xl,
                             const float* __restrict__ xr, const float* __restrict__ ee,
                             const float* __restrict__ att, float* __restrict__ logit,
                             unsigned* __restrict__ amax) {
  int wid = blockIdx.x * (blockDim.x >> 5) + (threadIdx.x >> 5);
  if (wid >= ETOT) return;
  int src, dst;
  if (wid < EEDG) { src = (int)ei[wid]; dst = (int)ei[EEDG + wid]; }
  else            { src = wid - EEDG;   dst = src; }
  int t = threadIdx.x & 31, f0 = t * 4, h = t >> 3, c0 = (t & 7) * 4;
  float4 a = *(const float4*)(xl + (long long)src * HCF + f0);
  float4 b = *(const float4*)(xr + (long long)dst * HCF + f0);
  float4 c = *(const float4*)(ee + (long long)wid * HCF + f0);
  float g[4] = {a.x + b.x + c.x, a.y + b.y + c.y, a.z + b.z + c.z, a.w + b.w + c.w};
  float s = 0.f;
#pragma unroll
  for (int j = 0; j < 4; ++j) {
    float gv = g[j];
    gv = gv > 0.f ? gv : NEG_SLOPE * gv;
    s += gv * att[h * NCH + c0 + j];
  }
  s += __shfl_xor(s, 1, 32);
  s += __shfl_xor(s, 2, 32);
  s += __shfl_xor(s, 4, 32);
  if ((t & 7) == 0) {
    logit[(long long)wid * NHEAD + h] = s;
    atomicMax(&amax[(long long)dst * NHEAD + h], f2ord(s));
  }
}

__global__ void edge_exp_k(const long long* __restrict__ ei, const float* __restrict__ logit,
                           const unsigned* __restrict__ amax, float* __restrict__ ex,
                           float* __restrict__ den) {
  long long i = (long long)blockIdx.x * blockDim.x + threadIdx.x;
  if (i >= (long long)ETOT * NHEAD) return;
  int e = (int)(i >> 2), h = (int)(i & 3);
  int dst = (e < EEDG) ? (int)ei[EEDG + e] : (e - EEDG);
  float v = __expf(logit[i] - ord2f(amax[(long long)dst * NHEAD + h]));
  ex[i] = v;
  atomicAdd(&den[(long long)dst * NHEAD + h], v);
}

__global__ void edge_agg_k(const long long* __restrict__ ei, const float* __restrict__ xl,
                           const float* __restrict__ ex, const float* __restrict__ den,
                           float* __restrict__ out) {
  int wid = blockIdx.x * (blockDim.x >> 5) + (threadIdx.x >> 5);
  if (wid >= ETOT) return;
  int src, dst;
  if (wid < EEDG) { src = (int)ei[wid]; dst = (int)ei[EEDG + wid]; }
  else            { src = wid - EEDG;   dst = src; }
  int t = threadIdx.x & 31, f0 = t * 4, h = t >> 3;
  float alpha = ex[(long long)wid * NHEAD + h] /
                (den[(long long)dst * NHEAD + h] + 1e-16f);
  float4 a = *(const float4*)(xl + (long long)src * HCF + f0);
  float* o = out + (long long)dst * HCF + f0;
  atomicAdd(o + 0, alpha * a.x);
  atomicAdd(o + 1, alpha * a.y);
  atomicAdd(o + 2, alpha * a.z);
  atomicAdd(o + 3, alpha * a.w);
}

__global__ void bias_relu_k(float* __restrict__ out, const float* __restrict__ bias,
                            long long total, int Nc) {
  long long i = (long long)blockIdx.x * blockDim.x + threadIdx.x;
  if (i >= total) return;
  int c = (int)(i % Nc);
  out[i] = fmaxf(out[i] + bias[c], 0.0f);
}

__global__ void head_k(const float* __restrict__ h2, const float* __restrict__ W3,
                       const float* __restrict__ b3, float* __restrict__ out) {
  int n = blockIdx.x * blockDim.x + threadIdx.x;
  if (n >= NN) return;
  float s = b3[0];
#pragma unroll 8
  for (int j = 0; j < 64; ++j) s += h2[(long long)n * 64 + j] * W3[j];
  out[n] = s;
}

// ---------- one GATv2 layer ----------
static void run_gat_layer(const float* xin, int Din,
                          const float* Wl, const float* bl, const float* Wr, const float* br,
                          const float* We, const float* att, const float* bias,
                          const long long* ei, const float* ea, const float* loopattr,
                          float* xl, float* xr, float* ee, float* logit, float* ex,
                          unsigned* amax, float* den, float* out, hipStream_t stream) {
  const int gN = (NN + 63) / 64;
  gemm_wmma_bf16<<<gN, 256, 0, stream>>>(xin, Wl, bl, xl, NN, Din, HCF, 0);
  gemm_wmma_bf16<<<gN, 256, 0, stream>>>(xin, Wr, br, xr, NN, Din, HCF, 0);
  gemm_wmma_bf16<<<(EEDG + 63) / 64, 256, 0, stream>>>(ea, We, nullptr, ee, EEDG, EDIM, HCF, 0);
  gemm_wmma_bf16<<<gN, 256, 0, stream>>>(loopattr, We, nullptr,
                                         ee + (long long)EEDG * HCF, NN, EDIM, HCF, 0);
  fill_u32<<<512, 256, 0, stream>>>(amax, ORD_NEG_INF, (long long)NN * NHEAD);
  fill_f32<<<512, 256, 0, stream>>>(den, 0.f, (long long)NN * NHEAD);
  fill_f32<<<4096, 256, 0, stream>>>(out, 0.f, (long long)NN * HCF);
  const int eb = (ETOT + 7) / 8;  // 8 waves per 256-thread block
  edge_logit_k<<<eb, 256, 0, stream>>>(ei, xl, xr, ee, att, logit, amax);
  edge_exp_k<<<(int)(((long long)ETOT * NHEAD + 255) / 256), 256, 0, stream>>>(ei, logit, amax, ex, den);
  edge_agg_k<<<eb, 256, 0, stream>>>(ei, xl, ex, den, out);
  bias_relu_k<<<(int)(((long long)NN * HCF + 255) / 256), 256, 0, stream>>>(out, bias,
                                                                            (long long)NN * HCF, HCF);
}

extern "C" void kernel_launch(void* const* d_in, const int* in_sizes, int n_in,
                              void* d_out, int out_size, void* d_ws, size_t ws_size,
                              hipStream_t stream) {
  const float*     x   = (const float*)d_in[0];
  const long long* ei  = (const long long*)d_in[1];   // int64 [2,E]
  const float*     ea  = (const float*)d_in[2];
  const float *c1_Wl = (const float*)d_in[3],  *c1_bl = (const float*)d_in[4];
  const float *c1_Wr = (const float*)d_in[5],  *c1_br = (const float*)d_in[6];
  const float *c1_We = (const float*)d_in[7],  *c1_att = (const float*)d_in[8];
  const float *c1_bias = (const float*)d_in[9];
  const float *c2_Wl = (const float*)d_in[10], *c2_bl = (const float*)d_in[11];
  const float *c2_Wr = (const float*)d_in[12], *c2_br = (const float*)d_in[13];
  const float *c2_We = (const float*)d_in[14], *c2_att = (const float*)d_in[15];
  const float *c2_bias = (const float*)d_in[16];
  const float *W1 = (const float*)d_in[17], *b1 = (const float*)d_in[18];
  const float *W2 = (const float*)d_in[19], *b2 = (const float*)d_in[20];
  const float *W3 = (const float*)d_in[21], *b3 = (const float*)d_in[22];

  // workspace carve-up (floats)
  float* ws = (float*)d_ws;
  size_t o = 0;
  float* cnt      = ws + o; o += NN;
  float* loopattr = ws + o; o += (size_t)NN * EDIM;
  float* xl       = ws + o; o += (size_t)NN * HCF;
  float* xr       = ws + o; o += (size_t)NN * HCF;
  float* ee       = ws + o; o += (size_t)ETOT * HCF;
  float* logit    = ws + o; o += (size_t)ETOT * NHEAD;
  float* ex       = ws + o; o += (size_t)ETOT * NHEAD;
  unsigned* amax  = (unsigned*)(ws + o); o += (size_t)NN * NHEAD;
  float* den      = ws + o; o += (size_t)NN * NHEAD;
  float* h1       = ws + o; o += (size_t)NN * HCF;   // layer1 out
  float* h2       = ws + o; o += (size_t)NN * HCF;   // layer2 out
  float* m1       = xl;                               // reuse: MLP hidden1 [N,128]
  float* m2       = xr;                               // reuse: MLP hidden2 [N,64]

  // self-loop edge_attr = per-dst mean of incoming edge_attr
  fill_f32<<<128, 256, 0, stream>>>(cnt, 0.f, NN);
  fill_f32<<<2048, 256, 0, stream>>>(loopattr, 0.f, (long long)NN * EDIM);
  deg_loopsum<<<EEDG, 128, 0, stream>>>(ei, ea, cnt, loopattr);
  loopattr_fin<<<NN, 128, 0, stream>>>(loopattr, cnt);

  run_gat_layer(x,  DIN, c1_Wl, c1_bl, c1_Wr, c1_br, c1_We, c1_att, c1_bias,
                ei, ea, loopattr, xl, xr, ee, logit, ex, amax, den, h1, stream);
  run_gat_layer(h1, HCF, c2_Wl, c2_bl, c2_Wr, c2_br, c2_We, c2_att, c2_bias,
                ei, ea, loopattr, xl, xr, ee, logit, ex, amax, den, h2, stream);

  // MLP head
  const int gN = (NN + 63) / 64;
  gemm_wmma_bf16<<<gN, 256, 0, stream>>>(h2, W1, b1, m1, NN, HCF, 128, 1);
  gemm_wmma_bf16<<<gN, 256, 0, stream>>>(m1, W2, b2, m2, NN, 128, 64, 1);
  head_k<<<(NN + 255) / 256, 256, 0, stream>>>(m2, W3, b3, (float*)d_out);

  (void)in_sizes; (void)n_in; (void)out_size; (void)ws_size;
}
